// MultiHeadAttention_16827681866045
// MI455X (gfx1250) — compile-verified, matrix operand order checked by reading, not run on hardware
//
#include <hip/hip_runtime.h>

// ---------------------------------------------------------------------------
// Flash-attention for B=4, H=16, S=1024, D=64 on gfx1250 (CDNA5, wave32).
// QK^T, P*ones (row sums) and P@V all run on v_wmma_f32_16x16x32_f16.
// Double-buffered LDS tiles, one barrier per key step, global loads issued
// ahead of compute, b128 LDS fragment loads, packed-f16 max butterflies,
// global_prefetch for tile n+2.
// ---------------------------------------------------------------------------

typedef __attribute__((ext_vector_type(16))) _Float16 v16h;
typedef __attribute__((ext_vector_type(2)))  _Float16 h2;
typedef __attribute__((ext_vector_type(8)))  float    v8f;

constexpr int SEQ   = 1024;   // sequence length S
constexpr int HD    = 64;     // head dim D
constexpr int KT    = 32;     // keys per flash step (one f16 WMMA K-dim)
constexpr int WAVES = 8;
constexpr int BLOCK = WAVES * 32;
constexpr int RPW   = 16;                 // query rows per wave
constexpr int RPB   = WAVES * RPW;        // 128 rows per block
constexpr int NSTEP = SEQ / KT;           // 32

union Frag { v16h h; uint4 q[2]; };
union H2U  { _Float16 h[2]; unsigned int u; };
union H2V  { h2 v; unsigned int u; };

// A/B 16-bit fragment from a row-major f16 row whose layout is contiguous in
// the WMMA K dimension: halves 0..7 = row[hi8 .. hi8+7],
// halves 8..15 = row[16+hi8 .. 16+hi8+7]  ->  two ds_load_b128 per lane.
__device__ __forceinline__ v16h ld_frag(const _Float16* row, int hi8) {
    Frag f;
    f.q[0] = *(const uint4*)(row + hi8);
    f.q[1] = *(const uint4*)(row + 16 + hi8);
    return f.h;
}

#define WMMA_F16(A, B, C) \
    __builtin_amdgcn_wmma_f32_16x16x32_f16(false, (A), false, (B), (short)0, (C), false, false)

__global__ __launch_bounds__(BLOCK)
void mha_flash_wmma_kernel(const float* __restrict__ q,     // [B*H, S, D]
                           const float* __restrict__ kT,    // [B*H, D, S]
                           const int*   __restrict__ scale_p,
                           const float* __restrict__ mask,  // [S, S]
                           const float* __restrict__ vmat,  // [B*H, S, D]
                           float* __restrict__ out)         // [B*H, S, D]
{
    const int head = blockIdx.y;
    const int tid  = threadIdx.x;
    const int wave = tid >> 5;
    const int lane = tid & 31;
    const int ncol = lane & 15;
    const int hi8  = (lane >> 4) << 3;     // 0 or 8
    const int qrow0 = blockIdx.x * RPB + wave * RPW;

    const float* qh = q    + (size_t)head * SEQ * HD;
    const float* kh = kT   + (size_t)head * HD  * SEQ;
    const float* vh = vmat + (size_t)head * SEQ * HD;
    float*       oh = out  + (size_t)head * SEQ * HD;
    const float inv_scale = 1.0f / (float)scale_p[0];

    // Double-buffered tiles. K: [key][d] (K-dim of QK^T frag = d, contiguous).
    // V: [d][sigma(key)]  (K-dim of PV frag = key, contiguous, sigma-ordered).
    __shared__ __align__(16) _Float16 Ksh[2][KT * HD];          // 8 KB
    __shared__ __align__(16) _Float16 Vsh[2][HD * KT];          // 8 KB
    __shared__ __align__(16) _Float16 Psh[WAVES][RPW * KT];     // 8 KB

    // ---- staging decomposition (one (2d x 4key) K unit + (2key x 4d) V unit
    //      per thread per tile; float4 global loads, b32 packed LDS stores) --
    const int kd0   = (tid >> 3) * 2;      // K: d pair base (0..62)
    const int kkey4 = (tid & 7) * 4;       // K: key group of 4
    const int vkey  = tid & 15;            // V: key (paired with vkey+16)
    const int vd4   = (tid >> 4) * 4;      // V: d group of 4

    float4 kr0, kr1, vr0, vr1;
    auto load_tile = [&](int kb) {
        kr0 = *(const float4*)(kh + (size_t)kd0 * SEQ + kb + kkey4);
        kr1 = *(const float4*)(kh + (size_t)(kd0 + 1) * SEQ + kb + kkey4);
        vr0 = *(const float4*)(vh + (size_t)(kb + vkey) * HD + vd4);
        vr1 = *(const float4*)(vh + (size_t)(kb + vkey + 16) * HD + vd4);
    };
    auto store_tile = [&](int buf) {
        const float* k0p = &kr0.x; const float* k1p = &kr1.x;
        const float* v0p = &vr0.x; const float* v1p = &vr1.x;
        #pragma unroll
        for (int j = 0; j < 4; ++j) {
            H2U a; a.h[0] = (_Float16)k0p[j]; a.h[1] = (_Float16)k1p[j];
            *(unsigned int*)(&Ksh[buf][(kkey4 + j) * HD + kd0]) = a.u;
            // physical column 2*vkey holds key vkey, 2*vkey+1 holds key vkey+16
            H2U b; b.h[0] = (_Float16)v0p[j]; b.h[1] = (_Float16)v1p[j];
            *(unsigned int*)(&Vsh[buf][(vd4 + j) * KT + 2 * vkey]) = b.u;
        }
    };

    // ---- Q A-fragments (16x32 f16, two covering D=64), built once ----------
    v16h a0, a1;
    {
        const float* qrow = qh + (size_t)(qrow0 + ncol) * HD;
        #pragma unroll
        for (int half = 0; half < 2; ++half) {          // halves 0..7 / 8..15
            const int dbase = half * 16 + hi8;
            float4 f0 = *(const float4*)(qrow + dbase);
            float4 f1 = *(const float4*)(qrow + dbase + 4);
            float4 g0 = *(const float4*)(qrow + 32 + dbase);
            float4 g1 = *(const float4*)(qrow + 32 + dbase + 4);
            const float* fp0 = &f0.x; const float* fp1 = &f1.x;
            const float* gp0 = &g0.x; const float* gp1 = &g1.x;
            #pragma unroll
            for (int j = 0; j < 4; ++j) {
                a0[half * 8 + j]     = (_Float16)fp0[j];
                a0[half * 8 + 4 + j] = (_Float16)fp1[j];
                a1[half * 8 + j]     = (_Float16)gp0[j];
                a1[half * 8 + 4 + j] = (_Float16)gp1[j];
            }
        }
    }

    v16h ones;
    #pragma unroll
    for (int i = 0; i < 16; ++i) ones[i] = (_Float16)1.0f;

    // ---- online-softmax state (replicated across each 16-lane half) --------
    float mrowv[8], lrowv[8];
    #pragma unroll
    for (int r = 0; r < 8; ++r) { mrowv[r] = -3.0e38f; lrowv[r] = 0.0f; }
    v8f o0 = {}, o1 = {}, o2 = {}, o3 = {};

    // ---- prologue: stage tile 0 --------------------------------------------
    load_tile(0);
    store_tile(0);
    __syncthreads();

    for (int step = 0; step < NSTEP; ++step) {
        const int kb  = step * KT;
        const int cur = step & 1;

        // issue next tile's global loads before compute (overlap with WMMA)
        if (step + 1 < NSTEP) load_tile(kb + KT);
        if (step + 2 < NSTEP) {
            __builtin_prefetch(kh + (size_t)kd0 * SEQ + kb + 2 * KT + kkey4, 0, 1);
            __builtin_prefetch(vh + (size_t)(kb + 2 * KT + vkey) * HD + vd4, 0, 1);
        }

        // ---- scores: S(16x32) = Q(16x64) x K^T, 4 WMMAs ----
        const _Float16* krow0 = &Ksh[cur][ncol * HD];          // key = ncol
        const _Float16* krow1 = &Ksh[cur][(16 + ncol) * HD];   // key = 16+ncol
        v16h bk00 = ld_frag(krow0, hi8);
        v16h bk01 = ld_frag(krow0 + 32, hi8);
        v16h bk10 = ld_frag(krow1, hi8);
        v16h bk11 = ld_frag(krow1 + 32, hi8);
        v8f c0 = {}, c1 = {};
        c0 = WMMA_F16(a0, bk00, c0);
        c0 = WMMA_F16(a1, bk01, c0);
        c1 = WMMA_F16(a0, bk10, c1);
        c1 = WMMA_F16(a1, bk11, c1);

        // ---- scale + mask ----
        float s0v[8], s1v[8];
        const float* mbase = mask + (size_t)(qrow0 + hi8) * SEQ + kb;
        #pragma unroll
        for (int r = 0; r < 8; ++r) {
            const float* mp = mbase + (size_t)r * SEQ;
            s0v[r] = c0[r] * inv_scale + mp[ncol];
            s1v[r] = c1[r] * inv_scale + mp[16 + ncol];
        }

        // ---- row max over 32 keys: two rows packed as 2xf16 per butterfly
        //      (v_pk_max_num_f16; f16 precision is fine for a stability max) --
        float mxv[8];
        #pragma unroll
        for (int rp = 0; rp < 4; ++rp) {
            H2V m2;
            m2.v[0] = (_Float16)fmaxf(s0v[2 * rp],     s1v[2 * rp]);
            m2.v[1] = (_Float16)fmaxf(s0v[2 * rp + 1], s1v[2 * rp + 1]);
            #pragma unroll
            for (int m = 1; m <= 8; m <<= 1) {
                H2V o2; o2.u = __shfl_xor(m2.u, m);
                m2.v = __builtin_elementwise_max(m2.v, o2.v);
            }
            mxv[2 * rp]     = (float)m2.v[0];
            mxv[2 * rp + 1] = (float)m2.v[1];
        }

        // ---- online softmax update + packed P spill ----
        float alpha[8];
        _Float16* pw = Psh[wave];
        #pragma unroll
        for (int r = 0; r < 8; ++r) {
            const float mnew = fmaxf(mrowv[r], mxv[r]);
            alpha[r] = __expf(mrowv[r] - mnew);
            mrowv[r] = mnew;
            const float p0 = __expf(s0v[r] - mnew);
            const float p1 = __expf(s1v[r] - mnew);
            // packed spill: physical cols (2n, 2n+1) = logical keys (n, 16+n)
            H2U pp; pp.h[0] = (_Float16)p0; pp.h[1] = (_Float16)p1;
            *(unsigned int*)(&pw[(r + hi8) * KT + 2 * ncol]) = pp.u;
        }
        #pragma unroll
        for (int r = 0; r < 8; ++r) {
            o0[r] *= alpha[r]; o1[r] *= alpha[r];
            o2[r] *= alpha[r]; o3[r] *= alpha[r];
        }

        // ---- reload P in A-layout (DS in-order within wave) ----
        v16h pa = ld_frag(&pw[ncol * KT], hi8);

        // ---- row sums of P via WMMA against all-ones B (replicated result) --
        v8f csum = {};
        csum = WMMA_F16(pa, ones, csum);
        #pragma unroll
        for (int r = 0; r < 8; ++r) lrowv[r] = lrowv[r] * alpha[r] + csum[r];

        // ---- PV: O(16x64) += P(16x32) x V(32x64), 4 WMMAs ----
        {
            v16h bv0 = ld_frag(&Vsh[cur][(0 * 16 + ncol) * KT], hi8);
            v16h bv1 = ld_frag(&Vsh[cur][(1 * 16 + ncol) * KT], hi8);
            v16h bv2 = ld_frag(&Vsh[cur][(2 * 16 + ncol) * KT], hi8);
            v16h bv3 = ld_frag(&Vsh[cur][(3 * 16 + ncol) * KT], hi8);
            o0 = WMMA_F16(pa, bv0, o0);
            o1 = WMMA_F16(pa, bv1, o1);
            o2 = WMMA_F16(pa, bv2, o2);
            o3 = WMMA_F16(pa, bv3, o3);
        }

        // stage next tile into the other buffer; one barrier per step
        if (step + 1 < NSTEP) store_tile(cur ^ 1);
        __syncthreads();
    }

    // ---- epilogue: normalize by row sum, store fp32 ------------------------
    #pragma unroll
    for (int r = 0; r < 8; ++r) {
        const int grow = qrow0 + r + hi8;
        const float invl = 1.0f / lrowv[r];
        oh[(size_t)grow * HD + 0 * 16 + ncol] = o0[r] * invl;
        oh[(size_t)grow * HD + 1 * 16 + ncol] = o1[r] * invl;
        oh[(size_t)grow * HD + 2 * 16 + ncol] = o2[r] * invl;
        oh[(size_t)grow * HD + 3 * 16 + ncol] = o3[r] * invl;
    }
}

extern "C" void kernel_launch(void* const* d_in, const int* in_sizes, int n_in,
                              void* d_out, int out_size, void* d_ws, size_t ws_size,
                              hipStream_t stream) {
    (void)in_sizes; (void)n_in; (void)out_size; (void)d_ws; (void)ws_size;

    const float* q       = (const float*)d_in[0];   // [4,16,1024,64]
    const float* kT      = (const float*)d_in[1];   // [4,16,64,1024]
    const int*   scale_p = (const int*)  d_in[2];   // scalar 8
    const float* mask    = (const float*)d_in[3];   // [1024,1024]
    const float* v       = (const float*)d_in[4];   // [4,16,1024,64]
    float*       out     = (float*)d_out;           // [4,16,1024,64]

    const int heads = 4 * 16;
    dim3 grid(SEQ / RPB, heads);                    // (8, 64)
    dim3 block(BLOCK);                              // 256 threads = 8 wave32
    mha_flash_wmma_kernel<<<grid, block, 0, stream>>>(q, kT, scale_p, mask, v, out);
}
